// Attention_40381282516996
// MI455X (gfx1250) — matrix operand order repaired and checked by reading, not compiled, not run on hardware
//
#include <hip/hip_runtime.h>

// ---------------------------------------------------------------------------
// MHA forward for gfx1250 (MI455X): bf16 WMMA (f32 accum) for all matmuls.
//   x[8,1024,768] @ w_qkv[768,2304] -> Q,K,V ; flash attention ; @ w_proj + b
// Round 2: software-pipelined GEMM K-loops (hide load latency behind WMMAs),
// attention K/V tiles staged via global_load_async_to_lds_b128 (double-
// buffered, s_wait_asynccnt), fragments served from LDS (ds_load_b128).
// ---------------------------------------------------------------------------

typedef __attribute__((ext_vector_type(16))) __bf16 v16bf;
typedef __attribute__((ext_vector_type(8)))  __bf16 v8bf;
typedef __attribute__((ext_vector_type(8)))  float  v8f;

#define DIM    768
#define HEADS  12
#define HD     64
#define BATCH  8
#define SEQ    1024
#define TOK    (BATCH * SEQ)   /* 8192 */
#define QKVC   (3 * DIM)       /* 2304 */
#define ASCALE 0.125f          /* 1/sqrt(64) */
#define KSTEPS (DIM / 32)      /* 24 */

static __device__ __forceinline__ v16bf cat16(v8bf lo, v8bf hi) {
  return __builtin_shufflevector(lo, hi, 0, 1, 2, 3, 4, 5, 6, 7,
                                         8, 9, 10, 11, 12, 13, 14, 15);
}

// A-fragment (16x32 bf16, M = lane&15): e<8 -> K = kb + 8*half + e,
//                                       e>=8 -> K = kb + 16 + 8*half + (e-8)
static __device__ __forceinline__ v16bf load_a_frag(const __bf16* rowp, int kb, int half) {
  v8bf lo = *(const v8bf*)(rowp + kb + 8 * half);
  v8bf hi = *(const v8bf*)(rowp + kb + 16 + 8 * half);
  return cat16(lo, hi);
}

// B-fragment (32x16 bf16, N = lane&15): e<8 -> K = kb + 8*half + e,
//                                       e>=8 -> K = kb + 16 + 8*half + (e-8)
static __device__ __forceinline__ v16bf load_b_frag(const __bf16* colp, int kb, int half) {
  v8bf lo = *(const v8bf*)(colp + kb + 8 * half);
  v8bf hi = *(const v8bf*)(colp + kb + 16 + 8 * half);
  return cat16(lo, hi);
}

static __device__ __forceinline__ v8f wmma_bf16(v16bf a, v16bf b, v8f c) {
  return __builtin_amdgcn_wmma_f32_16x16x32_bf16(false, a, false, b,
                                                 (short)0, c, false, false);
}

// gfx1250 async copy global -> LDS (16 bytes per active lane, ASYNCcnt).
static __device__ __forceinline__ void async_load_b128(unsigned lds_off, const void* gaddr) {
  asm volatile("global_load_async_to_lds_b128 %0, %1, off"
               :: "v"(lds_off), "v"(gaddr) : "memory");
}
#define WAIT_ASYNC_LE4() asm volatile("s_wait_asynccnt 0x4" ::: "memory")
#define WAIT_ASYNC_0()   asm volatile("s_wait_asynccnt 0x0" ::: "memory")

// ---------------------------------------------------------------------------
// Precision conversion kernels
// ---------------------------------------------------------------------------
__global__ void cvt_f32_bf16(const float* __restrict__ in, __bf16* __restrict__ out, int n) {
  int i = (blockIdx.x * blockDim.x + threadIdx.x) * 4;
  if (i < n) {
    float4 v = *(const float4*)(in + i);
    out[i + 0] = (__bf16)v.x;
    out[i + 1] = (__bf16)v.y;
    out[i + 2] = (__bf16)v.z;
    out[i + 3] = (__bf16)v.w;
  }
}

// out[c][r] = in[r][c]  (bf16), so B-fragments read contiguous K per column.
__global__ void cvtT_f32_bf16(const float* __restrict__ in, __bf16* __restrict__ out,
                              int rows, int cols) {
  int i = blockIdx.x * blockDim.x + threadIdx.x;
  if (i < rows * cols) {
    int r = i / cols, c = i % cols;
    out[(size_t)c * rows + r] = (__bf16)in[i];
  }
}

// ---------------------------------------------------------------------------
// QKV GEMM: xb[8192,768] @ wqkvT -> scatter to Q[N,d], K[N,d], Vt[d,N] (bf16)
// block = 128 threads (4 waves); block tile 64x64; wave tile 16x64.
// Software pipelined: next K-step's fragments load under current WMMAs.
// ---------------------------------------------------------------------------
__global__ __launch_bounds__(128)
void qkv_gemm(const __bf16* __restrict__ xb, const __bf16* __restrict__ wT,
              __bf16* __restrict__ Qb, __bf16* __restrict__ Kb, __bf16* __restrict__ Vt) {
  const int lane = threadIdx.x & 31;
  const int wave = threadIdx.x >> 5;
  const int half = lane >> 4;
  const int ln   = lane & 15;
  const int mbase = blockIdx.y * 64 + wave * 16;
  const int nbase = blockIdx.x * 64;

  v8f acc0 = {}, acc1 = {}, acc2 = {}, acc3 = {};
  const __bf16* arow  = xb + (size_t)(mbase + ln) * DIM;
  const __bf16* bcol0 = wT + (size_t)(nbase + 0 * 16 + ln) * DIM;
  const __bf16* bcol1 = wT + (size_t)(nbase + 1 * 16 + ln) * DIM;
  const __bf16* bcol2 = wT + (size_t)(nbase + 2 * 16 + ln) * DIM;
  const __bf16* bcol3 = wT + (size_t)(nbase + 3 * 16 + ln) * DIM;

  v16bf a  = load_a_frag(arow, 0, half);
  v16bf b0 = load_b_frag(bcol0, 0, half);
  v16bf b1 = load_b_frag(bcol1, 0, half);
  v16bf b2 = load_b_frag(bcol2, 0, half);
  v16bf b3 = load_b_frag(bcol3, 0, half);

  for (int ks = 0; ks < KSTEPS - 1; ++ks) {
    const int kn = (ks + 1) * 32;
    v16bf an = load_a_frag(arow, kn, half);
    v16bf n0 = load_b_frag(bcol0, kn, half);
    v16bf n1 = load_b_frag(bcol1, kn, half);
    v16bf n2 = load_b_frag(bcol2, kn, half);
    v16bf n3 = load_b_frag(bcol3, kn, half);
    __builtin_prefetch(arow + kn + 32, 0, 1);   // gfx1250 global_prefetch_b8
    acc0 = wmma_bf16(a, b0, acc0);
    acc1 = wmma_bf16(a, b1, acc1);
    acc2 = wmma_bf16(a, b2, acc2);
    acc3 = wmma_bf16(a, b3, acc3);
    a = an; b0 = n0; b1 = n1; b2 = n2; b3 = n3;
  }
  acc0 = wmma_bf16(a, b0, acc0);
  acc1 = wmma_bf16(a, b1, acc1);
  acc2 = wmma_bf16(a, b2, acc2);
  acc3 = wmma_bf16(a, b3, acc3);

  // Epilogue: D layout M = r + 8*half, N = ln. Scatter by (s, h, dd).
  v8f accs[4] = {acc0, acc1, acc2, acc3};
#pragma unroll
  for (int nt = 0; nt < 4; ++nt) {
    const int c   = nbase + nt * 16 + ln;
    const int s   = c / DIM;
    const int rem = c % DIM;
    const int h   = rem / HD;
    const int dd  = rem % HD;
#pragma unroll
    for (int r = 0; r < 8; ++r) {
      const int row = mbase + r + 8 * half;
      const int b   = row >> 10;
      const int tok = row & (SEQ - 1);
      const __bf16 val = (__bf16)accs[nt][r];
      const size_t bh = (size_t)(b * HEADS + h);
      if (s == 0)      Qb[(bh * SEQ + tok) * HD + dd] = val;
      else if (s == 1) Kb[(bh * SEQ + tok) * HD + dd] = val;
      else             Vt[(bh * HD + dd) * SEQ + tok] = val;   // V transposed
    }
  }
}

// ---------------------------------------------------------------------------
// Flash attention: one (b,h) per blockIdx.y, 64 query rows per block
// (16 per wave).  K/V tiles async-staged into LDS (double buffered) and
// shared by all 4 waves; online softmax; P restaged via LDS into A-layout.
// ---------------------------------------------------------------------------
__global__ __launch_bounds__(128)
void attn_kernel(const __bf16* __restrict__ Qb, const __bf16* __restrict__ Kb,
                 const __bf16* __restrict__ Vt, __bf16* __restrict__ attnb) {
  __shared__ __bf16 kls[2][32 * HD];    // K tile  [32 keys][64 d]   x2 buffers
  __shared__ __bf16 vls[2][HD * 32];    // Vt tile [64 d][32 keys]   x2 buffers
  __shared__ __bf16 plds[4][16 * 40];   // per-wave 16x32 P tile, padded stride

  const int tid  = threadIdx.x;
  const int lane = tid & 31;
  const int wave = tid >> 5;
  const int half = lane >> 4;
  const int ln   = lane & 15;
  const int bh = blockIdx.y;                 // 0..95
  const int b  = bh / HEADS, h = bh % HEADS;
  const int qbase = blockIdx.x * 64 + wave * 16;

  const __bf16* qp = Qb + (size_t)bh * SEQ * HD;
  const __bf16* kp = Kb + (size_t)bh * SEQ * HD;
  const __bf16* vp = Vt + (size_t)bh * HD * SEQ;

  // Q fragments: loaded once, reused for every key tile.
  const __bf16* qrow = qp + (size_t)(qbase + ln) * HD;
  const v16bf qa0 = load_a_frag(qrow, 0, half);
  const v16bf qa1 = load_a_frag(qrow, 32, half);

  v8f o0 = {}, o1 = {}, o2 = {}, o3 = {};
  float mrow[8], lrow[8];
#pragma unroll
  for (int r = 0; r < 8; ++r) { mrow[r] = -1e30f; lrow[r] = 0.0f; }

  __bf16* pl = plds[wave];
  const int NT = SEQ / 32;   // 32 key-pair steps

  // Stage a 32-key K/V tile pair into LDS buffer `buf` (4 async b128/thread).
  auto stage = [&](int key0, int buf) {
#pragma unroll
    for (int i = 0; i < 2; ++i) {               // K: 256 16B chunks, [32][64]
      const int c = tid + i * 128;
      const int row = c >> 3, col = (c & 7) * 8;
      async_load_b128((unsigned)(size_t)(&kls[buf][c * 8]),
                      kp + (size_t)(key0 + row) * HD + col);
    }
#pragma unroll
    for (int i = 0; i < 2; ++i) {               // V: 256 16B chunks, [64][32]
      const int c = tid + i * 128;
      const int row = c >> 2, col = (c & 3) * 8;
      async_load_b128((unsigned)(size_t)(&vls[buf][c * 8]),
                      vp + (size_t)row * SEQ + key0 + col);
    }
  };

  stage(0, 0);   // prime the pipeline

  for (int kt = 0; kt < NT; ++kt) {
    const int buf = kt & 1;
    if (kt + 1 < NT) {
      stage((kt + 1) * 32, buf ^ 1);   // 8 async ops in flight per wave
      WAIT_ASYNC_LE4();                // retire current buffer's 4 (in-order)
    } else {
      WAIT_ASYNC_0();
    }
    __syncthreads();                   // tile visible to all waves

    const __bf16* kb_ = kls[buf];
    const __bf16* vb_ = vls[buf];

    // S = Q @ K^T from LDS (contraction over d=64 -> 2 k-steps per key tile)
    v8f s0 = {}, s1 = {};
    s0 = wmma_bf16(qa0, load_b_frag(kb_ + (size_t)(0 * 16 + ln) * HD, 0, half), s0);
    s0 = wmma_bf16(qa1, load_b_frag(kb_ + (size_t)(0 * 16 + ln) * HD, 32, half), s0);
    s1 = wmma_bf16(qa0, load_b_frag(kb_ + (size_t)(1 * 16 + ln) * HD, 0, half), s1);
    s1 = wmma_bf16(qa1, load_b_frag(kb_ + (size_t)(1 * 16 + ln) * HD, 32, half), s1);

    // Online softmax over the 32 keys. Each row's 16 columns live across one
    // 16-lane group (lanes 0-15 hold rows r, lanes 16-31 rows r+8).
    float alpha[8], p0[8], p1[8];
#pragma unroll
    for (int r = 0; r < 8; ++r) {
      const float v0 = s0[r] * ASCALE, v1 = s1[r] * ASCALE;
      float t = fmaxf(v0, v1);
      for (int off = 1; off < 16; off <<= 1) t = fmaxf(t, __shfl_xor(t, off, 32));
      const float mn = fmaxf(mrow[r], t);
      alpha[r] = __expf(mrow[r] - mn);
      mrow[r] = mn;
      const float e0 = __expf(v0 - mn);
      const float e1 = __expf(v1 - mn);
      p0[r] = e0; p1[r] = e1;
      float ps = e0 + e1;
      for (int off = 1; off < 16; off <<= 1) ps += __shfl_xor(ps, off, 32);
      lrow[r] = alpha[r] * lrow[r] + ps;
    }

    // Rescale O and stage P (D layout) into LDS as a 16x32 row-major tile.
#pragma unroll
    for (int r = 0; r < 8; ++r) {
      o0[r] *= alpha[r]; o1[r] *= alpha[r]; o2[r] *= alpha[r]; o3[r] *= alpha[r];
      const int prow = r + 8 * half;
      pl[prow * 40 + ln]      = (__bf16)p0[r];
      pl[prow * 40 + 16 + ln] = (__bf16)p1[r];
    }
    __syncthreads();

    // Re-read P as a 16x32 A-fragment (M = ln, K = key-in-block).
    v16bf pa;
    {
      const __bf16* prow = pl + ln * 40;
      v8bf lo = *(const v8bf*)(prow + 8 * half);
      v8bf hi = *(const v8bf*)(prow + 16 + 8 * half);
      pa = cat16(lo, hi);
    }
    __syncthreads();

    // O += P @ V from LDS (B-fragment: contiguous keys per d column).
    o0 = wmma_bf16(pa, load_b_frag(vb_ + (size_t)(0 * 16 + ln) * 32, 0, half), o0);
    o1 = wmma_bf16(pa, load_b_frag(vb_ + (size_t)(1 * 16 + ln) * 32, 0, half), o1);
    o2 = wmma_bf16(pa, load_b_frag(vb_ + (size_t)(2 * 16 + ln) * 32, 0, half), o2);
    o3 = wmma_bf16(pa, load_b_frag(vb_ + (size_t)(3 * 16 + ln) * 32, 0, half), o3);
  }

  // Normalize by l and write bf16 attention output [8192, 768].
#pragma unroll
  for (int r = 0; r < 8; ++r) {
    const float inv = 1.0f / lrow[r];
    const int tok = qbase + r + 8 * half;
    const size_t orow = ((size_t)b * SEQ + tok) * DIM + (size_t)h * HD;
    attnb[orow + 0 * 16 + ln] = (__bf16)(o0[r] * inv);
    attnb[orow + 1 * 16 + ln] = (__bf16)(o1[r] * inv);
    attnb[orow + 2 * 16 + ln] = (__bf16)(o2[r] * inv);
    attnb[orow + 3 * 16 + ln] = (__bf16)(o3[r] * inv);
  }
}

// ---------------------------------------------------------------------------
// Projection GEMM: attnb[8192,768] @ wprojT + bias -> out fp32 (pipelined)
// ---------------------------------------------------------------------------
__global__ __launch_bounds__(128)
void proj_gemm(const __bf16* __restrict__ ab, const __bf16* __restrict__ wT,
               const float* __restrict__ bias, float* __restrict__ out) {
  const int lane = threadIdx.x & 31;
  const int wave = threadIdx.x >> 5;
  const int half = lane >> 4;
  const int ln   = lane & 15;
  const int mbase = blockIdx.y * 64 + wave * 16;
  const int nbase = blockIdx.x * 64;

  v8f acc0 = {}, acc1 = {}, acc2 = {}, acc3 = {};
  const __bf16* arow  = ab + (size_t)(mbase + ln) * DIM;
  const __bf16* bcol0 = wT + (size_t)(nbase + 0 * 16 + ln) * DIM;
  const __bf16* bcol1 = wT + (size_t)(nbase + 1 * 16 + ln) * DIM;
  const __bf16* bcol2 = wT + (size_t)(nbase + 2 * 16 + ln) * DIM;
  const __bf16* bcol3 = wT + (size_t)(nbase + 3 * 16 + ln) * DIM;

  v16bf a  = load_a_frag(arow, 0, half);
  v16bf b0 = load_b_frag(bcol0, 0, half);
  v16bf b1 = load_b_frag(bcol1, 0, half);
  v16bf b2 = load_b_frag(bcol2, 0, half);
  v16bf b3 = load_b_frag(bcol3, 0, half);

  for (int ks = 0; ks < KSTEPS - 1; ++ks) {
    const int kn = (ks + 1) * 32;
    v16bf an = load_a_frag(arow, kn, half);
    v16bf n0 = load_b_frag(bcol0, kn, half);
    v16bf n1 = load_b_frag(bcol1, kn, half);
    v16bf n2 = load_b_frag(bcol2, kn, half);
    v16bf n3 = load_b_frag(bcol3, kn, half);
    __builtin_prefetch(arow + kn + 32, 0, 1);
    acc0 = wmma_bf16(a, b0, acc0);
    acc1 = wmma_bf16(a, b1, acc1);
    acc2 = wmma_bf16(a, b2, acc2);
    acc3 = wmma_bf16(a, b3, acc3);
    a = an; b0 = n0; b1 = n1; b2 = n2; b3 = n3;
  }
  acc0 = wmma_bf16(a, b0, acc0);
  acc1 = wmma_bf16(a, b1, acc1);
  acc2 = wmma_bf16(a, b2, acc2);
  acc3 = wmma_bf16(a, b3, acc3);

  v8f accs[4] = {acc0, acc1, acc2, acc3};
#pragma unroll
  for (int nt = 0; nt < 4; ++nt) {
    const int c = nbase + nt * 16 + ln;
    const float bv = bias[c];
#pragma unroll
    for (int r = 0; r < 8; ++r) {
      const int row = mbase + r + 8 * half;
      out[(size_t)row * DIM + c] = accs[nt][r] + bv;
    }
  }
}

// ---------------------------------------------------------------------------
// Host-side launcher
// ---------------------------------------------------------------------------
extern "C" void kernel_launch(void* const* d_in, const int* in_sizes, int n_in,
                              void* d_out, int out_size, void* d_ws, size_t ws_size,
                              hipStream_t stream) {
  const float* x      = (const float*)d_in[0];   // [8,1024,768]
  const float* w_qkv  = (const float*)d_in[1];   // [768,2304]
  const float* w_proj = (const float*)d_in[2];   // [768,768]
  const float* b_proj = (const float*)d_in[3];   // [768]
  float* out = (float*)d_out;

  // Workspace layout (bf16), ~68 MB total.
  char* ws = (char*)d_ws;
  size_t off = 0;
  __bf16* xb     = (__bf16*)(ws + off); off += (size_t)TOK * DIM * 2;
  __bf16* wqkvT  = (__bf16*)(ws + off); off += (size_t)DIM * QKVC * 2;
  __bf16* wprojT = (__bf16*)(ws + off); off += (size_t)DIM * DIM * 2;
  __bf16* Qb     = (__bf16*)(ws + off); off += (size_t)TOK * DIM * 2;  // [96][1024][64]
  __bf16* Kb     = (__bf16*)(ws + off); off += (size_t)TOK * DIM * 2;  // [96][1024][64]
  __bf16* Vt     = (__bf16*)(ws + off); off += (size_t)TOK * DIM * 2;  // [96][64][1024]
  __bf16* attnb  = (__bf16*)(ws + off); off += (size_t)TOK * DIM * 2;  // [8192][768]

  // 1) Precision conversion (+ weight transposes for contiguous B-fragments).
  {
    int n = TOK * DIM;
    cvt_f32_bf16<<<n / (256 * 4), 256, 0, stream>>>(x, xb, n);
  }
  {
    int n = DIM * QKVC;
    cvtT_f32_bf16<<<(n + 255) / 256, 256, 0, stream>>>(w_qkv, wqkvT, DIM, QKVC);
  }
  {
    int n = DIM * DIM;
    cvtT_f32_bf16<<<(n + 255) / 256, 256, 0, stream>>>(w_proj, wprojT, DIM, DIM);
  }

  // 2) QKV GEMM (scatters into Q, K, V^T).
  qkv_gemm<<<dim3(QKVC / 64, TOK / 64), 128, 0, stream>>>(xb, wqkvT, Qb, Kb, Vt);

  // 3) Flash attention per (b, h).
  attn_kernel<<<dim3(SEQ / 64, BATCH * HEADS), 128, 0, stream>>>(Qb, Kb, Vt, attnb);

  // 4) Output projection + bias.
  proj_gemm<<<dim3(DIM / 64, TOK / 64), 128, 0, stream>>>(attnb, wprojT, b_proj, out);
}
